// Block_19164144075586
// MI455X (gfx1250) — matrix operand order repaired and checked by reading, not compiled, not run on hardware
//
#include <hip/hip_runtime.h>
#include <hip/hip_bf16.h>

typedef _Float16 h16_t;
typedef __attribute__((ext_vector_type(16))) _Float16 v16h;
typedef __attribute__((ext_vector_type(8)))  _Float16 v8h;
typedef __attribute__((ext_vector_type(8)))  float    v8f;

#define DIM   384
#define NH    8
#define HD    48
#define HDP   64          // HD padded to 64 for K=32 WMMA steps
#define HID   1536
#define NTOK  2048
#define BATCH 4
#define NROWS (BATCH*NTOK)   // 8192
#define LNEPS 1e-5f
#define ATT_SCALE 0.14433756729740643f  // 48^-0.5
#define SROW  2052           // padded LDS row stride (floats) for score strip
#define OPROW 52             // padded partial-O row stride

// ---------------- WMMA helpers (CDNA5 16x16x32 f16, f32 accum) ----------------

static __device__ inline v8f wmma_f16(v16h a, v16h b, v8f c) {
  return __builtin_amdgcn_wmma_f32_16x16x32_f16(false, a, false, b, (short)0, c,
                                                false, false);
}

// A-matrix 16x32 (MxK), row-major source with leading dim ld (elements).
// ISA layout: lane<16 -> M=lane, K = {k0..k0+7} in halves 0-7 and {k0+16..k0+23}
// in halves 8-15; lane>=16 -> M=lane-16, K = {k0+8..} / {k0+24..}.
static __device__ inline v16h loadA(const h16_t* __restrict__ rowbase, int ld,
                                    int k0, int lane) {
  int m  = lane & 15;
  int kb = k0 + ((lane >> 4) << 3);
  const h16_t* p = rowbase + (size_t)m * ld + kb;
  union { v16h v; v8h h[2]; } u;
  u.h[0] = *(const v8h*)(p);
  u.h[1] = *(const v8h*)(p + 16);
  return u.v;
}

// B-matrix 32x16 (KxN), source row-major [K,N] with leading dim ld.
// ISA layout: lane<16 -> N=lane, K = {0..7} in halves 0-7 and {16..23} in
// halves 8-15; lane>=16 -> N=lane-16, K = {8..15} / {24..31}.
static __device__ inline v16h loadB(const h16_t* __restrict__ base, int ld,
                                    int lane) {
  const h16_t* p = base + (size_t)((lane >> 4) << 3) * ld + (lane & 15);
  union { v16h v; _Float16 e[16]; } u;
#pragma unroll
  for (int i = 0; i < 8; ++i) {
    u.e[i] = p[(size_t)i * ld];
    u.e[8 + i] = p[(size_t)(16 + i) * ld];
  }
  return u.v;
}

// ---------------- prep kernels ----------------

// dst[k*dln + coff + n] = (f16) src[n*k_sz + k]   (weight transpose+cast)
__global__ void tcast_kernel(const float* __restrict__ src,
                             h16_t* __restrict__ dst, int n_sz, int k_sz,
                             int dln, int coff) {
  size_t total = (size_t)n_sz * k_sz;
  for (size_t idx = (size_t)blockIdx.x * blockDim.x + threadIdx.x; idx < total;
       idx += (size_t)gridDim.x * blockDim.x) {
    int nn = (int)(idx / k_sz);
    int kk = (int)(idx % k_sz);
    dst[(size_t)kk * dln + coff + nn] = (h16_t)src[idx];
  }
}

__global__ void fillz_kernel(h16_t* __restrict__ p, size_t n) {
  for (size_t i = (size_t)blockIdx.x * blockDim.x + threadIdx.x; i < n;
       i += (size_t)gridDim.x * blockDim.x)
    p[i] = (h16_t)0.0f;
}

// ---------------- layernorm (1 wave per 384-elem row) ----------------

__global__ void ln_kernel(const float* __restrict__ x,
                          const float* __restrict__ g,
                          const float* __restrict__ bb,
                          float* __restrict__ out32,
                          h16_t* __restrict__ out16) {
  int row  = blockIdx.x * 8 + (threadIdx.x >> 5);
  int lane = threadIdx.x & 31;
  const float* xr = x + (size_t)row * DIM;
  float v[12];
  float s = 0.f;
#pragma unroll
  for (int i = 0; i < 12; ++i) { v[i] = xr[i * 32 + lane]; s += v[i]; }
#pragma unroll
  for (int o = 16; o >= 1; o >>= 1) s += __shfl_xor(s, o, 32);
  float mu = s * (1.0f / DIM);
  float vs = 0.f;
#pragma unroll
  for (int i = 0; i < 12; ++i) { float d = v[i] - mu; vs += d * d; }
#pragma unroll
  for (int o = 16; o >= 1; o >>= 1) vs += __shfl_xor(vs, o, 32);
  float rstd = rsqrtf(vs * (1.0f / DIM) + LNEPS);
#pragma unroll
  for (int i = 0; i < 12; ++i) {
    int e = i * 32 + lane;
    float hn = (v[i] - mu) * rstd * g[e] + bb[e];
    if (out32) out32[(size_t)row * DIM + e] = hn;
    out16[(size_t)row * DIM + e] = (h16_t)hn;
  }
}

// ---------------- fused QKV GEMM + scatter epilogue ----------------
// C[row, n] = h16[row,:] . Wqkv_t[:, n],  n in [0,1152): 0-383 q, 384-767 k, 768+ v
__global__ void qkv_kernel(const h16_t* __restrict__ A,
                           const h16_t* __restrict__ Bt,
                           h16_t* __restrict__ q16, h16_t* __restrict__ kT16,
                           h16_t* __restrict__ v16) {
  const int N = 3 * DIM, K = DIM;
  int wid  = (int)((blockIdx.x * (size_t)blockDim.x + threadIdx.x) >> 5);
  int lane = threadIdx.x & 31;
  const int nt64 = N >> 6;
  if (wid >= (NROWS >> 4) * nt64) return;
  int m0 = (wid / nt64) << 4;
  int n0 = (wid % nt64) << 6;
  v8f acc[4] = {};
  const h16_t* Arow = A + (size_t)m0 * K;
  for (int k0 = 0; k0 < K; k0 += 32) {
    v16h a = loadA(Arow, K, k0, lane);
    const h16_t* Bb = Bt + (size_t)k0 * N + n0;
#pragma unroll
    for (int s = 0; s < 4; ++s)
      acc[s] = wmma_f16(a, loadB(Bb + s * 16, N, lane), acc[s]);
  }
  int rb = (lane >> 4) << 3;
#pragma unroll
  for (int s = 0; s < 4; ++s) {
    int n = n0 + s * 16 + (lane & 15);
    int region = n / DIM, ci = n % DIM;
    int hh = ci / HD, d = ci % HD;
#pragma unroll
    for (int r = 0; r < 8; ++r) {
      int row = m0 + r + rb;
      int b = row >> 11, tok = row & (NTOK - 1);
      size_t bh = (size_t)b * NH + hh;
      float c = acc[s][r];
      if (region == 0)
        q16[(bh * NTOK + tok) * HDP + d] = (h16_t)(c * ATT_SCALE);
      else if (region == 1)
        kT16[(bh * HDP + d) * NTOK + tok] = (h16_t)c;
      else
        v16[(bh * NTOK + tok) * HD + d] = (h16_t)c;
    }
  }
}

// ---------------- fused attention: S = qk^T, softmax, write probs, O = P v ---
// grid = (NTOK/16, BATCH*NH); block = 256 (8 waves); dyn LDS = score strip + partials
__global__ void attn_kernel(const h16_t* __restrict__ q16,
                            const h16_t* __restrict__ kT16,
                            const h16_t* __restrict__ v16,
                            float* __restrict__ attn_out,
                            h16_t* __restrict__ o16) {
  extern __shared__ float smem[];
  float* S  = smem;                 // [16][SROW]
  float* OP = smem + 16 * SROW;     // [8][16][OPROW]
  int bh = blockIdx.y;              // b*NH + h
  int m0 = blockIdx.x * 16;
  int wave = threadIdx.x >> 5, lane = threadIdx.x & 31;
  int rb = (lane >> 4) << 3;

  // ---- scores: each wave covers 256 key columns ----
  const h16_t* qb = q16 + ((size_t)bh * NTOK + m0) * HDP;
  const h16_t* kb = kT16 + (size_t)bh * HDP * NTOK;
  v16h a0 = loadA(qb, HDP, 0, lane);
  v16h a1 = loadA(qb, HDP, 32, lane);
  for (int t = 0; t < 16; ++t) {
    int key0 = wave * 256 + t * 16;
    v8f c = {};
    c = wmma_f16(a0, loadB(kb + key0, NTOK, lane), c);
    c = wmma_f16(a1, loadB(kb + (size_t)32 * NTOK + key0, NTOK, lane), c);
    int col = key0 + (lane & 15);
#pragma unroll
    for (int r = 0; r < 8; ++r) S[(r + rb) * SROW + col] = c[r];
  }
  __syncthreads();

  // ---- softmax: wave handles rows 2w, 2w+1; write normalized probs ----
  for (int rr = 0; rr < 2; ++rr) {
    int row = wave * 2 + rr;
    float* srow = S + (size_t)row * SROW;
    float mx = -3.4e38f;
    for (int i = lane; i < NTOK; i += 32) mx = fmaxf(mx, srow[i]);
#pragma unroll
    for (int o = 16; o >= 1; o >>= 1) mx = fmaxf(mx, __shfl_xor(mx, o, 32));
    float sm = 0.f;
    for (int i = lane; i < NTOK; i += 32) {
      float e = __expf(srow[i] - mx);
      srow[i] = e;
      sm += e;
    }
#pragma unroll
    for (int o = 16; o >= 1; o >>= 1) sm += __shfl_xor(sm, o, 32);
    float inv = 1.0f / sm;
    float* grow = attn_out + ((size_t)bh * NTOK + m0 + row) * NTOK;
    for (int i = lane; i < NTOK; i += 32) {
      float p = srow[i] * inv;
      srow[i] = p;
      grow[i] = p;
    }
  }
  __syncthreads();

  // ---- O = P * V : wave accumulates its 256-key slice, 3 hd tiles ----
  v8f oacc[3] = {};
  const h16_t* vb = v16 + (size_t)bh * NTOK * HD;
  for (int j = 0; j < 8; ++j) {
    int key0 = wave * 256 + j * 32;
    int m = lane & 15;
    int kb2 = key0 + rb;  // rb doubles as (lane>=16 ? 8 : 0)
    union { v16h v; _Float16 e[16]; } a;
#pragma unroll
    for (int h = 0; h < 8; ++h) a.e[h] = (h16_t)S[m * SROW + kb2 + h];
#pragma unroll
    for (int h = 0; h < 8; ++h) a.e[8 + h] = (h16_t)S[m * SROW + kb2 + 16 + h];
#pragma unroll
    for (int s = 0; s < 3; ++s)
      oacc[s] =
          wmma_f16(a.v, loadB(vb + (size_t)key0 * HD + s * 16, HD, lane), oacc[s]);
  }
#pragma unroll
  for (int s = 0; s < 3; ++s) {
    int n = s * 16 + (lane & 15);
#pragma unroll
    for (int r = 0; r < 8; ++r)
      OP[(wave * 16 + r + rb) * OPROW + n] = oacc[s][r];
  }
  __syncthreads();

  // ---- reduce 8 partials, store o16 ----
  int b = bh >> 3, hh = bh & 7;
  for (int i = threadIdx.x; i < 16 * HD; i += 256) {
    int mrow = i / HD, n = i % HD;
    float sum = 0.f;
#pragma unroll
    for (int w = 0; w < 8; ++w) sum += OP[(w * 16 + mrow) * OPROW + n];
    size_t row = (size_t)b * NTOK + m0 + mrow;
    o16[row * DIM + hh * HD + n] = (h16_t)sum;
  }
}

// ---------------- generic WMMA GEMM with epilogue ----------------
// mode 0: out32 = resid + (C + bias)      (f32)
// mode 1: out16 = gelu(C + bias)          (f16, exact erf GELU)
__global__ void gemm_kernel(const h16_t* __restrict__ A,
                            const h16_t* __restrict__ Bt,
                            const float* __restrict__ bias,
                            const float* __restrict__ resid,
                            float* __restrict__ out32,
                            h16_t* __restrict__ out16, int M, int N, int K,
                            int mode) {
  int wid  = (int)((blockIdx.x * (size_t)blockDim.x + threadIdx.x) >> 5);
  int lane = threadIdx.x & 31;
  int nt64 = N >> 6;
  if (wid >= (M >> 4) * nt64) return;
  int m0 = (wid / nt64) << 4;
  int n0 = (wid % nt64) << 6;
  v8f acc[4] = {};
  const h16_t* Arow = A + (size_t)m0 * K;
  for (int k0 = 0; k0 < K; k0 += 32) {
    v16h a = loadA(Arow, K, k0, lane);
    const h16_t* Bb = Bt + (size_t)k0 * N + n0;
#pragma unroll
    for (int s = 0; s < 4; ++s)
      acc[s] = wmma_f16(a, loadB(Bb + s * 16, N, lane), acc[s]);
  }
  int rb = (lane >> 4) << 3;
#pragma unroll
  for (int s = 0; s < 4; ++s) {
    int n = n0 + s * 16 + (lane & 15);
    float bv = bias ? bias[n] : 0.f;
#pragma unroll
    for (int r = 0; r < 8; ++r) {
      int row = m0 + r + rb;
      float c = acc[s][r] + bv;
      size_t o = (size_t)row * N + n;
      if (mode == 0)
        out32[o] = resid[o] + c;
      else
        out16[o] = (h16_t)(0.5f * c * (1.0f + erff(c * 0.70710678118654752f)));
    }
  }
}

// ---------------- host launcher ----------------

extern "C" void kernel_launch(void* const* d_in, const int* in_sizes, int n_in,
                              void* d_out, int out_size, void* d_ws,
                              size_t ws_size, hipStream_t stream) {
  const float* x     = (const float*)d_in[0];
  const float* wq    = (const float*)d_in[1];
  const float* wk    = (const float*)d_in[2];
  const float* wv    = (const float*)d_in[3];
  const float* wo    = (const float*)d_in[4];
  const float* bo    = (const float*)d_in[5];
  const float* g1    = (const float*)d_in[6];
  const float* b1    = (const float*)d_in[7];
  const float* g2    = (const float*)d_in[8];
  const float* b2    = (const float*)d_in[9];
  const float* wfc1  = (const float*)d_in[10];
  const float* bfc1  = (const float*)d_in[11];
  const float* wfc2  = (const float*)d_in[12];
  const float* bfc2  = (const float*)d_in[13];

  float* y_out    = (float*)d_out;
  float* attn_out = y_out + (size_t)NROWS * DIM;

  // workspace carve (256B aligned)
  char* w = (char*)d_ws;
  auto carve = [&](size_t bytes) {
    char* p = w;
    w += (bytes + 255) & ~(size_t)255;
    return p;
  };
  float* h32   = (float*)carve((size_t)NROWS * DIM * 4);
  float* yw    = (float*)carve((size_t)NROWS * DIM * 4);
  h16_t* h16   = (h16_t*)carve((size_t)NROWS * DIM * 2);
  h16_t* q16   = (h16_t*)carve((size_t)BATCH * NH * NTOK * HDP * 2);
  h16_t* kT16  = (h16_t*)carve((size_t)BATCH * NH * NTOK * HDP * 2);
  h16_t* v16   = (h16_t*)carve((size_t)BATCH * NH * NTOK * HD * 2);
  h16_t* o16   = (h16_t*)carve((size_t)NROWS * DIM * 2);
  h16_t* z16   = (h16_t*)carve((size_t)NROWS * DIM * 2);
  h16_t* a16   = (h16_t*)carve((size_t)NROWS * HID * 2);
  h16_t* Wqkv  = (h16_t*)carve((size_t)DIM * 3 * DIM * 2);
  h16_t* Wot   = (h16_t*)carve((size_t)DIM * DIM * 2);
  h16_t* Wf1t  = (h16_t*)carve((size_t)DIM * HID * 2);
  h16_t* Wf2t  = (h16_t*)carve((size_t)HID * DIM * 2);

  // 1) weight transpose + cast to f16 [K, N] layouts
  tcast_kernel<<<576, 256, 0, stream>>>(wq, Wqkv, DIM, DIM, 3 * DIM, 0);
  tcast_kernel<<<576, 256, 0, stream>>>(wk, Wqkv, DIM, DIM, 3 * DIM, DIM);
  tcast_kernel<<<576, 256, 0, stream>>>(wv, Wqkv, DIM, DIM, 3 * DIM, 2 * DIM);
  tcast_kernel<<<576, 256, 0, stream>>>(wo, Wot, DIM, DIM, DIM, 0);
  tcast_kernel<<<2048, 256, 0, stream>>>(wfc1, Wf1t, HID, DIM, HID, 0);
  tcast_kernel<<<2048, 256, 0, stream>>>(wfc2, Wf2t, DIM, HID, DIM, 0);

  // 2) zero q16 / kT16 (pad lanes 48..63 must be exactly 0)
  {
    size_t nq = (size_t)BATCH * NH * NTOK * HDP;
    fillz_kernel<<<2048, 256, 0, stream>>>(q16, nq);
    fillz_kernel<<<2048, 256, 0, stream>>>(kT16, nq);
  }

  // 3) LN1 -> h32 (residual source) + h16 (GEMM input)
  ln_kernel<<<NROWS / 8, 256, 0, stream>>>(x, g1, b1, h32, h16);

  // 4) QKV projection (M=8192, N=1152, K=384), scatter to q16/kT16/v16
  {
    int waves = (NROWS / 16) * ((3 * DIM) / 64);
    qkv_kernel<<<(waves * 32 + 255) / 256, 256, 0, stream>>>(h16, Wqkv, q16,
                                                             kT16, v16);
  }

  // 5) fused attention: probs -> d_out, O -> o16
  {
    dim3 g(NTOK / 16, BATCH * NH);
    size_t lds = (size_t)(16 * SROW + 8 * 16 * OPROW) * sizeof(float);
    attn_kernel<<<g, 256, lds, stream>>>(q16, kT16, v16, attn_out, o16);
  }

  // 6) O projection + bias + residual(h) -> yw  (M=8192,N=384,K=384)
  {
    int waves = (NROWS / 16) * (DIM / 64);
    gemm_kernel<<<(waves * 32 + 255) / 256, 256, 0, stream>>>(
        o16, Wot, bo, h32, yw, (h16_t*)nullptr, NROWS, DIM, DIM, 0);
  }

  // 7) LN2 -> z16
  ln_kernel<<<NROWS / 8, 256, 0, stream>>>(yw, g2, b2, (float*)nullptr, z16);

  // 8) FC1 + GELU -> a16  (M=8192,N=1536,K=384)
  {
    int waves = (NROWS / 16) * (HID / 64);
    gemm_kernel<<<(waves * 32 + 255) / 256, 256, 0, stream>>>(
        z16, Wf1t, bfc1, (const float*)nullptr, (float*)nullptr, a16, NROWS,
        HID, DIM, 1);
  }

  // 9) FC2 + bias + residual(yw) -> y_out  (M=8192,N=384,K=1536)
  {
    int waves = (NROWS / 16) * (DIM / 64);
    gemm_kernel<<<(waves * 32 + 255) / 256, 256, 0, stream>>>(
        a16, Wf2t, bfc2, yw, y_out, (h16_t*)nullptr, NROWS, DIM, HID, 0);
  }
}
